// PointNet2SSGFeature_59579786330794
// MI455X (gfx1250) — matrix operand order repaired and checked by reading, not compiled, not run on hardware
//
#include <hip/hip_runtime.h>
#include <hip/hip_bf16.h>

#define DEVINL __device__ __forceinline__

typedef __attribute__((ext_vector_type(16))) _Float16 v16h;
typedef __attribute__((ext_vector_type(8)))  float    v8f;

// ---------------------------------------------------------------------------
// WMMA helpers (gfx1250, wave32).  D = A(16x32 f16) * B(32x16 f16) + C(16x16 f32)
// A lane layout used here (contiguous k, not the ISA table order; 16-bit A 16x32):
//   lane: m=lane&15, half=lane>>4;  elem e: K = half*16 + e
// B lane layout (16-bit B 32x16):
//   lane: n=lane&15, half=lane>>4;  elem e: K = half*16 + e
// C/D layout: VGPR v: lanes0-15 -> (M=v, N=lane); lanes16-31 -> (M=v+8, N=lane-16)
//
// Weights are pre-swizzled (f16) so an A operand is ONE aligned v16h load.
// LDS activations are written pre-swizzled so a B operand is ONE v16h LDS load.
// ---------------------------------------------------------------------------

DEVINL v8f vzero8() {
  v8f a;
#pragma unroll
  for (int i = 0; i < 8; ++i) a[i] = 0.f;
  return a;
}

DEVINL v8f wmma_fma(v16h a, v16h b, v8f c) {
  return __builtin_amdgcn_wmma_f32_16x16x32_f16(false, a, false, b, (short)0, c,
                                                false, false);
}

// A operand: pre-swizzled f16 weights. Tile stride = 32 lanes * 16 halves = 512.
DEVINL v16h load_a_swz(const _Float16* __restrict__ Wswz, int kcTiles, int oT,
                       int kc) {
  const int lane = threadIdx.x & 31;
  return *(const v16h*)(Wswz + ((((size_t)oT * kcTiles) + kc) << 9) +
                        ((size_t)lane << 4));
}

// B operand: pre-swizzled f16 LDS activations. Chunk = (kc * nTiles + nT).
DEVINL v16h load_b_lds_swz(const _Float16* H, int nTiles, int kc, int nT) {
  const int lane = threadIdx.x & 31;
  return *(const v16h*)(H + (((size_t)kc * nTiles + nT) << 9) +
                        ((size_t)lane << 4));
}

// Swizzled LDS index for logical element (k, col) of a [K][cols] B matrix.
// Consumer lane = ((k>>4)&1)*16 + (col&15), element = k&15.
DEVINL int lds_swz_idx(int k, int col, int nTiles) {
  return ((((k >> 5) * nTiles) + (col >> 4)) << 9) +
         ((((k >> 4) & 1) << 4) + (col & 15)) * 16 + (k & 15);
}

// B tile from a global fp32 activation matrix X[C][M] (k bounds-checked).
DEVINL v16h load_b_global(const float* __restrict__ X, int C, int M, int kBase,
                          int colBase) {
  const int lane = threadIdx.x & 31;
  const int col  = colBase + (lane & 15);
  const int kb   = kBase + ((lane >> 4) << 4);
  v16h b;
#pragma unroll
  for (int j = 0; j < 8; ++j) {
    const int k0 = kb + 2 * j;
    b[2 * j]     = (_Float16)((k0     < C) ? X[(size_t)k0 * M + col]       : 0.f);
    b[2 * j + 1] = (_Float16)((k0 + 1 < C) ? X[(size_t)(k0 + 1) * M + col] : 0.f);
  }
  return b;
}

// relu(c*gamma+beta) -> swizzled f16 LDS matrix (B-operand layout for next layer)
DEVINL void bn_relu_to_lds_swz(v8f c, const float* __restrict__ gamma,
                               const float* __restrict__ beta, int oBase,
                               _Float16* H, int nTiles, int colBase) {
  const int lane = threadIdx.x & 31;
  const int half = lane >> 4;
  const int col  = colBase + (lane & 15);
#pragma unroll
  for (int v = 0; v < 8; ++v) {
    const int o = oBase + v + (half << 3);
    float y = c[v] * gamma[o] + beta[o];
    y = y > 0.f ? y : 0.f;
    H[lds_swz_idx(o, col, nTiles)] = (_Float16)y;
  }
}

// relu(c*gamma+beta), then max over the 16 columns of this tile (width-16
// butterfly within each half-wave), write per-(o, ntile) partial max.
DEVINL void bn_relu_rowmax(v8f c, const float* __restrict__ gamma,
                           const float* __restrict__ beta, int oBase,
                           float* partmax, int nTiles, int nT) {
  const int lane = threadIdx.x & 31;
  const int half = lane >> 4;
#pragma unroll
  for (int v = 0; v < 8; ++v) {
    const int o = oBase + v + (half << 3);
    float y = c[v] * gamma[o] + beta[o];
    y = y > 0.f ? y : 0.f;
#pragma unroll
    for (int off = 8; off > 0; off >>= 1)
      y = fmaxf(y, __shfl_xor(y, off, 16));
    if ((lane & 15) == 0) partmax[o * nTiles + nT] = y;
  }
}

// ---------------------------------------------------------------------------
// Pre-swizzle one fp32 weight matrix W[O][C] into the WMMA A-operand f16
// layout: out[((oT*kcTiles + kc)*32 + lane)*16 + e], zero-padded past C.
// ---------------------------------------------------------------------------
__global__ void swizzle_weights_kernel(const float* __restrict__ W,
                                       _Float16* __restrict__ out, int C, int O) {
  const int kcT   = (C + 31) >> 5;
  const int total = (O >> 4) * kcT * 512;
  for (int t = blockIdx.x * blockDim.x + threadIdx.x; t < total;
       t += gridDim.x * blockDim.x) {
    const int e    = t & 15;
    const int lane = (t >> 4) & 31;
    const int tile = t >> 9;
    const int kc   = tile % kcT;
    const int oT   = tile / kcT;
    const int half = lane >> 4;
    const int j    = e >> 1;
    const int k = (kc << 5) + (half << 4) + e;
    const int o = (oT << 4) + (lane & 15);
    out[t] = (_Float16)((k < C) ? W[(size_t)o * C + k] : 0.f);
  }
}

// ---------------------------------------------------------------------------
// Farthest point sampling: one block per batch. dist[] lives in LDS; 511/127
// sequential argmax rounds (latency-bound, tiny vs. GEMM work). Writes the
// selected centroid coordinates channel-major: cent[b*centStrideB + c*ncent + j].
// ---------------------------------------------------------------------------
__global__ void __launch_bounds__(256)
fps_centroids_kernel(const float* __restrict__ xyz, int n, int ncent,
                     float* __restrict__ cent, unsigned long long centStrideB) {
  __shared__ float dist[8192];
  __shared__ float rmax[256];
  __shared__ int   rarg[256];
  __shared__ int   sIdx[512];
  __shared__ int   sLast;
  const int b = blockIdx.x, tid = threadIdx.x;
  const float* px = xyz + (size_t)b * 3u * (size_t)n;
  const float* py = px + n;
  const float* pz = px + 2 * n;
  for (int i = tid; i < n; i += 256) dist[i] = 1e10f;
  if (tid == 0) { sIdx[0] = 0; sLast = 0; }
  __syncthreads();
  int last = 0;
  for (int it = 1; it < ncent; ++it) {
    const float cx = px[last], cy = py[last], cz = pz[last];
    float best = -1.f;
    int   bi   = 0;
    for (int i = tid; i < n; i += 256) {
      const float dx = px[i] - cx, dy = py[i] - cy, dz = pz[i] - cz;
      const float nd = fminf(dist[i], dx * dx + dy * dy + dz * dz);
      dist[i] = nd;
      if (nd > best) { best = nd; bi = i; }
    }
    rmax[tid] = best;
    rarg[tid] = bi;
    __syncthreads();
    for (int s = 128; s > 0; s >>= 1) {
      if (tid < s && rmax[tid + s] > rmax[tid]) {
        rmax[tid] = rmax[tid + s];
        rarg[tid] = rarg[tid + s];
      }
      __syncthreads();
    }
    if (tid == 0) { sLast = rarg[0]; sIdx[it] = sLast; }
    __syncthreads();
    last = sLast;
  }
  float* cb = cent + (size_t)b * centStrideB;
  for (int j = tid; j < ncent; j += 256) {
    const int idx = sIdx[j];
    cb[0 * ncent + j] = px[idx];
    cb[1 * ncent + j] = py[idx];
    cb[2 * ncent + j] = pz[idx];
  }
}

// ---------------------------------------------------------------------------
// SA0 fused kernel: one block (8 waves) per (centroid, batch).
//   wave0: in-index-order ball query over 8192 pts (wave32 ballot scan)
//   gather 3x32 relative xyz -> swizzled LDS f16 (K padded to 32)
//   MLP 3->64->64->128 via WMMA f16, BN+ReLU fused, intermediates f16 in LDS
//   max over k=32 neighbours -> feat0[b][128][512]
// ---------------------------------------------------------------------------
__global__ void __launch_bounds__(256)
sa0_group_mlp_kernel(const float* __restrict__ pts,      // [B][3][8192]
                     const float* __restrict__ cent,     // [B][3][512]
                     const _Float16* __restrict__ W0, const float* __restrict__ G0,
                     const float* __restrict__ Bt0,
                     const _Float16* __restrict__ W1, const float* __restrict__ G1,
                     const float* __restrict__ Bt1,
                     const _Float16* __restrict__ W2, const float* __restrict__ G2,
                     const float* __restrict__ Bt2,
                     float* __restrict__ outFeat)        // [B][128][512]
{
  constexpr int   N  = 8192, NC = 512, K = 32, NT = 2;  // NT = 32 cols / 16
  constexpr float R2 = 0.2f * 0.2f;
  __shared__ int      nIdx[K];
  __shared__ float    sc[3];
  __shared__ __align__(32) _Float16 Xin[32 * 32];   // swizzled [1 kc][2 nT]
  __shared__ __align__(32) _Float16 H1[64 * 32];    // swizzled [2 kc][2 nT]
  __shared__ __align__(32) _Float16 H2[64 * 32];
  __shared__ float    partmax[128 * 2];

  const int j = blockIdx.x, b = blockIdx.y, tid = threadIdx.x;
  const float* px = pts + (size_t)b * 3u * N;
  const float* py = px + N;
  const float* pz = px + 2 * N;
  if (tid < 3) sc[tid] = cent[((size_t)b * 3 + tid) * NC + j];
  __syncthreads();

  if (tid < 32) {  // wave0: ball query, first-K in index order
    const int   lane = tid;
    const float cx = sc[0], cy = sc[1], cz = sc[2];
    int cnt = 0;
    for (int base = 0; base < N && cnt < K; base += 32) {
      const int   i  = base + lane;
      const float dx = px[i] - cx, dy = py[i] - cy, dz = pz[i] - cz;
      const bool  in = (dx * dx + dy * dy + dz * dz) < R2;
      const unsigned m32 = __builtin_amdgcn_ballot_w32(in);
      const int pos = cnt + __popc(m32 & ((1u << lane) - 1u));
      if (in && pos < K) nIdx[pos] = i;
      cnt += __popc(m32);
      if (cnt > K) cnt = K;
    }
    for (int p = cnt + lane; p < K; p += 32) nIdx[p] = nIdx[0];  // pad w/ first
  }
  __syncthreads();

  for (int e = tid; e < 32 * 32; e += 256) {  // gather rel xyz, zero-pad K
    const int k = e >> 5, col = e & 31;
    float v = 0.f;
    if (k < 3) v = px[(size_t)k * N + nIdx[col]] - sc[k];
    Xin[lds_swz_idx(k, col, NT)] = (_Float16)v;
  }
  __syncthreads();

  const int w = tid >> 5;
  {  // layer1: 3(->32) -> 64 ; 4 o-tiles x 2 n-tiles = 8 wave-tasks
    const int oT = w >> 1, nT = w & 1;
    v8f acc = vzero8();
    acc = wmma_fma(load_a_swz(W0, 1, oT, 0), load_b_lds_swz(Xin, NT, 0, nT), acc);
    bn_relu_to_lds_swz(acc, G0, Bt0, oT * 16, H1, NT, nT * 16);
  }
  __syncthreads();
  {  // layer2: 64 -> 64
    const int oT = w >> 1, nT = w & 1;
    v8f acc = vzero8();
#pragma unroll
    for (int kc = 0; kc < 2; ++kc)
      acc = wmma_fma(load_a_swz(W1, 2, oT, kc), load_b_lds_swz(H1, NT, kc, nT),
                     acc);
    bn_relu_to_lds_swz(acc, G1, Bt1, oT * 16, H2, NT, nT * 16);
  }
  __syncthreads();
  // layer3: 64 -> 128, fused k-max (8 o-tiles x 2 n-tiles; wave w = o-tile w)
#pragma unroll
  for (int nT = 0; nT < 2; ++nT) {
    v8f acc = vzero8();
#pragma unroll
    for (int kc = 0; kc < 2; ++kc)
      acc = wmma_fma(load_a_swz(W2, 2, w, kc), load_b_lds_swz(H2, NT, kc, nT),
                     acc);
    bn_relu_rowmax(acc, G2, Bt2, w * 16, partmax, 2, nT);
  }
  __syncthreads();
  if (tid < 128)
    outFeat[((size_t)b * 128 + tid) * NC + j] =
        fmaxf(partmax[tid * 2 + 0], partmax[tid * 2 + 1]);
}

// ---------------------------------------------------------------------------
// SA1 fused kernel: block (8 waves) per (centroid, batch); 512 pts, k=64,
// channels 131 (3 xyz + 128 feat, K padded to 160) -> 128 -> 128 -> 256.
// Writes max-pooled features straight into rows 3..258 of the concat buffer
// X2[b][259][128] (rows 0..2 = centroid xyz written by FPS#2).
// ---------------------------------------------------------------------------
__global__ void __launch_bounds__(256)
sa1_group_mlp_kernel(const float* __restrict__ xyz,      // cent0 [B][3][512]
                     const float* __restrict__ feat,     // feat0 [B][128][512]
                     const _Float16* __restrict__ W0, const float* __restrict__ G0,
                     const float* __restrict__ Bt0,
                     const _Float16* __restrict__ W1, const float* __restrict__ G1,
                     const float* __restrict__ Bt1,
                     const _Float16* __restrict__ W2, const float* __restrict__ G2,
                     const float* __restrict__ Bt2,
                     float* __restrict__ outX2)          // [B][259][128]
{
  constexpr int   N = 512, NC = 128, K = 64, CIN = 131, CPAD = 160, NT = 4;
  constexpr float R2 = 0.4f * 0.4f;
  __shared__ int      nIdx[K];
  __shared__ float    sc[3];
  __shared__ __align__(32) _Float16 Xin[CPAD * 64];   // 20 KB, swizzled [5][4]
  __shared__ __align__(32) _Float16 H1[128 * 64];     // 16 KB, swizzled [4][4]
  __shared__ __align__(32) _Float16 H2[128 * 64];     // 16 KB
  __shared__ float    partmax[256 * 4];               //  4 KB

  const int j = blockIdx.x, b = blockIdx.y, tid = threadIdx.x;
  const float* px = xyz + (size_t)b * 3u * N;
  const float* py = px + N;
  const float* pz = px + 2 * N;
  if (tid < 3) sc[tid] = xyz[((size_t)b * 3 + tid) * N + j];
  __syncthreads();

  if (tid < 32) {  // ball query over 512 points
    const int   lane = tid;
    const float cx = sc[0], cy = sc[1], cz = sc[2];
    int cnt = 0;
    for (int base = 0; base < N && cnt < K; base += 32) {
      const int   i  = base + lane;
      const float dx = px[i] - cx, dy = py[i] - cy, dz = pz[i] - cz;
      const bool  in = (dx * dx + dy * dy + dz * dz) < R2;
      const unsigned m32 = __builtin_amdgcn_ballot_w32(in);
      const int pos = cnt + __popc(m32 & ((1u << lane) - 1u));
      if (in && pos < K) nIdx[pos] = i;
      cnt += __popc(m32);
      if (cnt > K) cnt = K;
    }
    for (int p = cnt + lane; p < K; p += 32) nIdx[p] = nIdx[0];
  }
  __syncthreads();

  for (int e = tid; e < CPAD * 64; e += 256) {  // gather rel-xyz + features
    const int k = e >> 6, col = e & 63;
    const int idx = nIdx[col];
    float v = 0.f;
    if (k < 3)        v = px[(size_t)k * N + idx] - sc[k];
    else if (k < CIN) v = feat[((size_t)b * 128 + (k - 3)) * N + idx];
    Xin[lds_swz_idx(k, col, NT)] = (_Float16)v;
  }
  __syncthreads();

  const int w = tid >> 5;
  // layer1: 131(->160) -> 128 ; 8 o-tiles x 4 n-tiles = 32 tasks, 4 per wave
#pragma unroll
  for (int i = 0; i < 4; ++i) {
    const int t = w + 8 * i, oT = t >> 2, nT = t & 3;
    v8f acc = vzero8();
#pragma unroll
    for (int kc = 0; kc < 5; ++kc)
      acc = wmma_fma(load_a_swz(W0, 5, oT, kc), load_b_lds_swz(Xin, NT, kc, nT),
                     acc);
    bn_relu_to_lds_swz(acc, G0, Bt0, oT * 16, H1, NT, nT * 16);
  }
  __syncthreads();
  // layer2: 128 -> 128
#pragma unroll
  for (int i = 0; i < 4; ++i) {
    const int t = w + 8 * i, oT = t >> 2, nT = t & 3;
    v8f acc = vzero8();
#pragma unroll
    for (int kc = 0; kc < 4; ++kc)
      acc = wmma_fma(load_a_swz(W1, 4, oT, kc), load_b_lds_swz(H1, NT, kc, nT),
                     acc);
    bn_relu_to_lds_swz(acc, G1, Bt1, oT * 16, H2, NT, nT * 16);
  }
  __syncthreads();
  // layer3: 128 -> 256 with fused k-max ; 16 o-tiles x 4 n-tiles = 64 tasks
#pragma unroll
  for (int i = 0; i < 8; ++i) {
    const int t = w + 8 * i, oT = t >> 2, nT = t & 3;
    v8f acc = vzero8();
#pragma unroll
    for (int kc = 0; kc < 4; ++kc)
      acc = wmma_fma(load_a_swz(W2, 4, oT, kc), load_b_lds_swz(H2, NT, kc, nT),
                     acc);
    bn_relu_rowmax(acc, G2, Bt2, oT * 16, partmax, 4, nT);
  }
  __syncthreads();
  {
    const float m = fmaxf(fmaxf(partmax[tid * 4 + 0], partmax[tid * 4 + 1]),
                          fmaxf(partmax[tid * 4 + 2], partmax[tid * 4 + 3]));
    outX2[(size_t)b * 259 * NC + (size_t)(3 + tid) * NC + j] = m;
  }
}

// ---------------------------------------------------------------------------
// Generic 1x1-conv / FC layer: Y[b][o][m] = relu(gamma[o]*(W X)[o][m] + beta[o])
// One wave per 16x16 tile; K-loop of v_wmma_f32_16x16x32_f16 with a
// global_prefetch on the next pre-swizzled weight tile.
// ---------------------------------------------------------------------------
__global__ void __launch_bounds__(32)
gemm_bn_relu_kernel(const float* __restrict__ X, const _Float16* __restrict__ Wswz,
                    const float* __restrict__ gamma, const float* __restrict__ beta,
                    float* __restrict__ Y, int C, int M,
                    unsigned long long strideXb, unsigned long long strideYb) {
  const int b = blockIdx.z;
  const float* Xb = X + (size_t)b * strideXb;
  float*       Yb = Y + (size_t)b * strideYb;
  const int oT = blockIdx.y, mBase = blockIdx.x * 16;
  const int kcT = (C + 31) >> 5;
  v8f acc = vzero8();
  for (int kc = 0; kc < kcT; ++kc) {
    if (kc + 1 < kcT)
      __builtin_prefetch(Wswz + ((((size_t)oT * kcT) + kc + 1) << 9), 0, 3);
    acc = wmma_fma(load_a_swz(Wswz, kcT, oT, kc),
                   load_b_global(Xb, C, M, kc * 32, mBase), acc);
  }
  const int lane = threadIdx.x & 31;
  const int half = lane >> 4;
  const int col  = mBase + (lane & 15);
#pragma unroll
  for (int v = 0; v < 8; ++v) {
    const int o = oT * 16 + v + (half << 3);
    float y = acc[v] * gamma[o] + beta[o];
    y = y > 0.f ? y : 0.f;
    Yb[(size_t)o * M + col] = y;
  }
}

// Global max pool over columns: Y[c*Bn + b] = max_m X[b][c][m]  (output is the
// transposed [C][B] layout the FC GEMM consumes directly as its B-operand).
__global__ void maxpool_cols_kernel(const float* __restrict__ X,
                                    float* __restrict__ Y, int C, int M,
                                    unsigned long long strideXb, int Bn) {
  const int tg = blockIdx.x * blockDim.x + threadIdx.x;
  if (tg >= C * Bn) return;
  const int c = tg / Bn, b = tg % Bn;
  const float* row = X + (size_t)b * strideXb + (size_t)c * M;
  float m = row[0];
  for (int i = 1; i < M; ++i) m = fmaxf(m, row[i]);
  Y[(size_t)c * Bn + b] = m;
}

// Final [O][B] -> [B][O]
__global__ void transpose_out_kernel(const float* __restrict__ X,
                                     float* __restrict__ out, int O, int Bn) {
  const int tg = blockIdx.x * blockDim.x + threadIdx.x;
  if (tg >= O * Bn) return;
  const int b = tg / O, o = tg % O;
  out[(size_t)b * O + o] = X[(size_t)o * Bn + b];
}

// ---------------------------------------------------------------------------
extern "C" void kernel_launch(void* const* d_in, const int* in_sizes, int n_in,
                              void* d_out, int out_size, void* d_ws,
                              size_t ws_size, hipStream_t stream) {
  (void)in_sizes; (void)n_in; (void)out_size; (void)ws_size;
  int ii = 0;
  const float* point = (const float*)d_in[ii++];
  struct Layer { const float *w, *g, *b; };
  Layer sa0[3], sa1[3], loc[3], glob[2];
  auto take = [&](Layer& L) {
    L.w = (const float*)d_in[ii++];
    L.g = (const float*)d_in[ii++];
    L.b = (const float*)d_in[ii++];
  };
  for (int i = 0; i < 3; ++i) take(sa0[i]);
  for (int i = 0; i < 3; ++i) take(sa1[i]);
  for (int i = 0; i < 3; ++i) take(loc[i]);
  for (int i = 0; i < 2; ++i) take(glob[i]);

  constexpr int Bn = 16;
  float* wsf = (float*)d_ws;
  size_t off = 0;
  float* cent0 = wsf + off; off += (size_t)Bn * 3 * 512;     // SA0 centroids
  float* feat0 = wsf + off; off += (size_t)Bn * 128 * 512;   // SA0 features
  float* X2    = wsf + off; off += (size_t)Bn * 259 * 128;   // [xyz1 ; feat1]
  float* L1    = wsf + off; off += (size_t)Bn * 256 * 128;
  float* L2o   = wsf + off; off += (size_t)Bn * 512 * 128;
  float* L3o   = wsf + off; off += (size_t)Bn * 1024 * 128;
  float* gl    = wsf + off; off += (size_t)1024 * Bn;        // [1024][B]
  float* fc1   = wsf + off; off += (size_t)512 * Bn;
  float* fc2   = wsf + off; off += (size_t)256 * Bn;

  // f16 pre-swizzled weight region (32B-aligned; each matrix a multiple of 1KB)
  off = (off + 7) & ~(size_t)7;
  _Float16* wsh  = (_Float16*)(wsf + off);
  auto swsz = [](int O, int C) {
    return (size_t)(O / 16) * (size_t)((C + 31) / 32) * 512;
  };
  size_t h = 0;
  _Float16* Wsa0[3]; _Float16* Wsa1[3]; _Float16* Wloc[3]; _Float16* Wglob[2];
  const int sa0OC[3][2]  = {{64, 3}, {64, 64}, {128, 64}};
  const int sa1OC[3][2]  = {{128, 131}, {128, 128}, {256, 128}};
  const int locOC[3][2]  = {{256, 259}, {512, 256}, {1024, 512}};
  const int globOC[2][2] = {{512, 1024}, {256, 512}};
  for (int i = 0; i < 3; ++i) { Wsa0[i] = wsh + h; h += swsz(sa0OC[i][0], sa0OC[i][1]); }
  for (int i = 0; i < 3; ++i) { Wsa1[i] = wsh + h; h += swsz(sa1OC[i][0], sa1OC[i][1]); }
  for (int i = 0; i < 3; ++i) { Wloc[i] = wsh + h; h += swsz(locOC[i][0], locOC[i][1]); }
  for (int i = 0; i < 2; ++i) { Wglob[i] = wsh + h; h += swsz(globOC[i][0], globOC[i][1]); }

  auto launch_swz = [&](const float* W, _Float16* out, int O, int C) {
    const int total = (O >> 4) * ((C + 31) >> 5) * 512;
    swizzle_weights_kernel<<<dim3((total + 255) / 256), 256, 0, stream>>>(W, out,
                                                                          C, O);
  };
  for (int i = 0; i < 3; ++i) launch_swz(sa0[i].w, Wsa0[i], sa0OC[i][0], sa0OC[i][1]);
  for (int i = 0; i < 3; ++i) launch_swz(sa1[i].w, Wsa1[i], sa1OC[i][0], sa1OC[i][1]);
  for (int i = 0; i < 3; ++i) launch_swz(loc[i].w, Wloc[i], locOC[i][0], locOC[i][1]);
  for (int i = 0; i < 2; ++i) launch_swz(glob[i].w, Wglob[i], globOC[i][0], globOC[i][1]);

  // SA0: FPS(8192 -> 512) then fused group+MLP+maxpool
  fps_centroids_kernel<<<Bn, 256, 0, stream>>>(point, 8192, 512, cent0,
                                               3ull * 512);
  sa0_group_mlp_kernel<<<dim3(512, Bn), 256, 0, stream>>>(
      point, cent0, Wsa0[0], sa0[0].g, sa0[0].b, Wsa0[1], sa0[1].g, sa0[1].b,
      Wsa0[2], sa0[2].g, sa0[2].b, feat0);

  // SA1: FPS(512 -> 128) writes xyz1 into X2 rows 0..2; fused MLP fills 3..258
  fps_centroids_kernel<<<Bn, 256, 0, stream>>>(cent0, 512, 128, X2,
                                               259ull * 128);
  sa1_group_mlp_kernel<<<dim3(128, Bn), 256, 0, stream>>>(
      cent0, feat0, Wsa1[0], sa1[0].g, sa1[0].b, Wsa1[1], sa1[1].g, sa1[1].b,
      Wsa1[2], sa1[2].g, sa1[2].b, X2);

  // Local MLP 259 -> 256 -> 512 -> 1024 over 128 points
  gemm_bn_relu_kernel<<<dim3(8, 16, Bn), 32, 0, stream>>>(
      X2, Wloc[0], loc[0].g, loc[0].b, L1, 259, 128, 259ull * 128, 256ull * 128);
  gemm_bn_relu_kernel<<<dim3(8, 32, Bn), 32, 0, stream>>>(
      L1, Wloc[1], loc[1].g, loc[1].b, L2o, 256, 128, 256ull * 128, 512ull * 128);
  gemm_bn_relu_kernel<<<dim3(8, 64, Bn), 32, 0, stream>>>(
      L2o, Wloc[2], loc[2].g, loc[2].b, L3o, 512, 128, 512ull * 128,
      1024ull * 128);

  // Global max pool -> [1024][B], then FC 1024 -> 512 -> 256 (M = B = 16)
  maxpool_cols_kernel<<<(1024 * Bn + 255) / 256, 256, 0, stream>>>(
      L3o, gl, 1024, 128, 1024ull * 128, Bn);
  gemm_bn_relu_kernel<<<dim3(1, 32, 1), 32, 0, stream>>>(
      gl, Wglob[0], glob[0].g, glob[0].b, fc1, 1024, Bn, 0ull, 0ull);
  gemm_bn_relu_kernel<<<dim3(1, 16, 1), 32, 0, stream>>>(
      fc1, Wglob[1], glob[1].g, glob[1].b, fc2, 512, Bn, 0ull, 0ull);

  transpose_out_kernel<<<(256 * Bn + 255) / 256, 256, 0, stream>>>(
      fc2, (float*)d_out, 256, Bn);
}
